// MLPPool_55430847922218
// MI455X (gfx1250) — matrix operand order repaired and checked by reading, not compiled, not run on hardware
//
#include <hip/hip_runtime.h>

// ---------------------------------------------------------------------------
// MLPPool on MI455X (gfx1250): Linear(768->512) + masked BatchNorm + ReLU
// + [pool-before-GEMM2 algebraic rewrite] + Linear(512->512) + bias.
// Heavy GEMM done with V_WMMA_F32_16X16X32_BF16 using bf16 hi/lo (bf16x3)
// split for fp32-class accuracy. All reductions are fixed-order (no float
// atomics) so results are bitwise deterministic across replays.
// ---------------------------------------------------------------------------

#define Bb   32
#define Ll   2048
#define Dk   768
#define Hn   512
#define Mtot (Bb * Ll)          // 65536 rows
#define EPSV 1e-5f

typedef __attribute__((ext_vector_type(16))) __bf16 v16bf;
typedef __attribute__((ext_vector_type(8)))  __bf16 v8bf;
typedef __attribute__((ext_vector_type(8)))  float  v8f;
typedef __attribute__((ext_vector_type(4)))  float  f32x4;
typedef __attribute__((ext_vector_type(2)))  float  f32x2;

static __device__ __forceinline__ v16bf cat16(v8bf a, v8bf b) {
    v16bf r;
#pragma unroll
    for (int i = 0; i < 8; ++i) { r[i] = a[i]; r[i + 8] = b[i]; }
    return r;
}

// ---------------------------------------------------------------------------
// Kernel 0: per-batch valid-token counts (exact integer reduction)
// ---------------------------------------------------------------------------
__global__ __launch_bounds__(256)
void k_counts(const int* __restrict__ attn, float* __restrict__ count) {
    __shared__ int red[256];
    const int b = blockIdx.x, t = threadIdx.x;
    int s = 0;
#pragma unroll
    for (int j = 0; j < Ll / 256; ++j) s += attn[b * Ll + t + 256 * j];
    red[t] = s;
    __syncthreads();
    for (int off = 128; off > 0; off >>= 1) {
        if (t < off) red[t] += red[t + off];
        __syncthreads();
    }
    if (t == 0) count[b] = (float)red[0];
}

// ---------------------------------------------------------------------------
// Kernel 1: h = hidden @ W1 + b1   (bf16x3 WMMA), store h, emit per-block
// masked sum / sumsq partials for the BatchNorm statistics.
// Block tile 128(M) x 128(N); 8 waves, each wave 32x64 (2x4 WMMA tiles).
// ---------------------------------------------------------------------------
__global__ __launch_bounds__(256)
void k_gemm1(const float* __restrict__ hidden, const int* __restrict__ attn,
             const float* __restrict__ W1, const float* __restrict__ b1,
             float* __restrict__ h_out,
             float* __restrict__ s1part, float* __restrict__ s2part) {
    __shared__ __attribute__((aligned(16))) __bf16 Ah[128][32];
    __shared__ __attribute__((aligned(16))) __bf16 Al[128][32];
    __shared__ __attribute__((aligned(16))) __bf16 Bh[128][32];
    __shared__ __attribute__((aligned(16))) __bf16 Bl[128][32];
    __shared__ float blk1[4][128];
    __shared__ float blk2[4][128];

    const int tid  = threadIdx.x;
    const int row0 = blockIdx.x * 128;   // 512 m-blocks
    const int col0 = blockIdx.y * 128;   // 4 n-blocks
    const int lane = tid & 31;
    const int wave = tid >> 5;
    const int wm   = wave & 3;           // 4 m-subtiles of 32 rows
    const int wn   = wave >> 2;          // 2 n-subtiles of 64 cols
    const int half = lane >> 4;          // lane group (ISA K-split)
    const int l16  = lane & 15;

    v8f acc[2][4];
#pragma unroll
    for (int mt = 0; mt < 2; ++mt)
#pragma unroll
        for (int nt = 0; nt < 4; ++nt)
#pragma unroll
            for (int i = 0; i < 8; ++i) acc[mt][nt][i] = 0.0f;

    for (int kk = 0; kk < Dk / 32; ++kk) {
        const int k0 = kk * 32;
        __syncthreads();   // previous iteration's WMMA reads done

        // ---- load A tile (128 rows x 32 k), fp32 -> bf16 hi/lo, NT loads ----
#pragma unroll
        for (int ii = 0; ii < 4; ++ii) {
            const int i  = tid + 256 * ii;
            const int r  = i >> 3;            // 8 float4 per row
            const int kp = (i & 7) << 2;
            const f32x4 v = __builtin_nontemporal_load(
                (const f32x4*)(hidden + (size_t)(row0 + r) * Dk + k0 + kp));
            if (ii == 0 && kk + 1 < Dk / 32)  // gfx1250 global_prefetch_b8
                __builtin_prefetch(hidden + (size_t)(row0 + r) * Dk + k0 + 32 + kp, 0, 1);
            float xs[4] = {v.x, v.y, v.z, v.w};
#pragma unroll
            for (int j = 0; j < 4; ++j) {
                const __bf16 hi = (__bf16)xs[j];
                Ah[r][kp + j] = hi;
                Al[r][kp + j] = (__bf16)(xs[j] - (float)hi);
            }
        }
        // ---- load B tile (32 k x 128 n) transposed into LDS [n][k] ----
#pragma unroll
        for (int ii = 0; ii < 4; ++ii) {
            const int i  = tid + 256 * ii;
            const int k  = i >> 5;            // 32 float4 per k-row
            const int np = (i & 31) << 2;
            const f32x4 v = *(const f32x4*)(W1 + (size_t)(k0 + k) * Hn + col0 + np);
            float xs[4] = {v.x, v.y, v.z, v.w};
#pragma unroll
            for (int j = 0; j < 4; ++j) {
                const __bf16 hi = (__bf16)xs[j];
                Bh[np + j][k] = hi;
                Bl[np + j][k] = (__bf16)(xs[j] - (float)hi);
            }
        }
        __syncthreads();

        // ---- fragments per ISA 7.12.2 layouts ----
        // A 16x32 bf16: lane group g, element j: k = 8g+j (j<8), 8g+8+j (j>=8)
        v16bf ah[2], al[2], bh[4], bl[4];
#pragma unroll
        for (int mt = 0; mt < 2; ++mt) {
            const int ar = wm * 32 + mt * 16 + l16;
            ah[mt] = cat16(*(const v8bf*)&Ah[ar][8 * half],
                           *(const v8bf*)&Ah[ar][8 * half + 16]);
            al[mt] = cat16(*(const v8bf*)&Al[ar][8 * half],
                           *(const v8bf*)&Al[ar][8 * half + 16]);
        }
        // B 32x16 bf16: lane group g, element j: k = 8g+j (j<8), 8g+8+j (j>=8)
#pragma unroll
        for (int nt = 0; nt < 4; ++nt) {
            const int br = wn * 64 + nt * 16 + l16;
            bh[nt] = cat16(*(const v8bf*)&Bh[br][8 * half],
                           *(const v8bf*)&Bh[br][8 * half + 16]);
            bl[nt] = cat16(*(const v8bf*)&Bl[br][8 * half],
                           *(const v8bf*)&Bl[br][8 * half + 16]);
        }
        // ---- bf16x3: hi*hi + hi*lo + lo*hi, fp32 accumulate ----
#pragma unroll
        for (int mt = 0; mt < 2; ++mt)
#pragma unroll
            for (int nt = 0; nt < 4; ++nt) {
                acc[mt][nt] = __builtin_amdgcn_wmma_f32_16x16x32_bf16(
                    false, ah[mt], false, bh[nt], (short)0, acc[mt][nt], false, false);
                acc[mt][nt] = __builtin_amdgcn_wmma_f32_16x16x32_bf16(
                    false, ah[mt], false, bl[nt], (short)0, acc[mt][nt], false, false);
                acc[mt][nt] = __builtin_amdgcn_wmma_f32_16x16x32_bf16(
                    false, al[mt], false, bh[nt], (short)0, acc[mt][nt], false, false);
            }
    }

    // ---- epilogue: + bias, store h, masked sum/sumsq partials -------------
    float mk[2][8];
#pragma unroll
    for (int mt = 0; mt < 2; ++mt)
#pragma unroll
        for (int r = 0; r < 8; ++r)
            mk[mt][r] = (float)attn[row0 + wm * 32 + mt * 16 + r + 8 * half];

#pragma unroll
    for (int nt = 0; nt < 4; ++nt) {
        const int colg = col0 + wn * 64 + nt * 16 + l16;
        const float bias = b1[colg];
        float p1 = 0.f, p2 = 0.f;
#pragma unroll
        for (int mt = 0; mt < 2; ++mt) {
            const int rowbase = row0 + wm * 32 + mt * 16 + 8 * half;
#pragma unroll
            for (int r = 0; r < 8; ++r) {   // C layout: VGPR r -> M = r + 8*half
                const float v = acc[mt][nt][r] + bias;
                h_out[(size_t)(rowbase + r) * Hn + colg] = v;
                const float m = mk[mt][r];
                p1 += m * v;
                p2 += m * v * v;
            }
        }
        p1 += __shfl_xor(p1, 16);   // lanes l and l+16 hold the same column
        p2 += __shfl_xor(p2, 16);
        if (half == 0) {            // each wave owns blk[wm][wn*64 .. +63]
            blk1[wm][wn * 64 + nt * 16 + l16] = p1;
            blk2[wm][wn * 64 + nt * 16 + l16] = p2;
        }
    }
    __syncthreads();
    if (tid < 128) {   // fixed-order combine -> deterministic partials
        const float t1 = blk1[0][tid] + blk1[1][tid] + blk1[2][tid] + blk1[3][tid];
        const float t2 = blk2[0][tid] + blk2[1][tid] + blk2[2][tid] + blk2[3][tid];
        s1part[(size_t)blockIdx.x * Hn + col0 + tid] = t1;
        s2part[(size_t)blockIdx.x * Hn + col0 + tid] = t2;
    }
}

// ---------------------------------------------------------------------------
// Kernel 2: fold 512 per-block partials per feature -> scale/shift
// ---------------------------------------------------------------------------
__global__ __launch_bounds__(512)
void k_stats(const float* __restrict__ s1part, const float* __restrict__ s2part,
             const float* __restrict__ count,
             const float* __restrict__ gamma, const float* __restrict__ beta,
             float* __restrict__ scale, float* __restrict__ shift) {
    const int n = threadIdx.x;   // 512 features
    float nv = 0.f;
    for (int b = 0; b < Bb; ++b) nv += count[b];
    nv = fmaxf(nv, 1.0f);
    float s1 = 0.f, s2 = 0.f;
    for (int mb = 0; mb < Mtot / 128; ++mb) {
        s1 += s1part[(size_t)mb * Hn + n];
        s2 += s2part[(size_t)mb * Hn + n];
    }
    const float mean = s1 / nv;
    const float var  = s2 / nv - mean * mean;
    const float sc   = gamma[n] * rsqrtf(var + EPSV);
    scale[n] = sc;
    shift[n] = beta[n] - mean * sc;
}

// ---------------------------------------------------------------------------
// Kernel 3: masked Sum_l relu(h*scale+shift) -> 16 deterministic partials/b
// ---------------------------------------------------------------------------
__global__ __launch_bounds__(256)
void k_pool(const float* __restrict__ h, const int* __restrict__ attn,
            const float* __restrict__ scale, const float* __restrict__ shift,
            float* __restrict__ poolpart) {
    const int b = blockIdx.y;        // 32
    const int c = blockIdx.x;        // 16 chunks of 128 rows
    const int n = threadIdx.x * 2;   // two features per thread
    const float sc0 = scale[n], sc1 = scale[n + 1];
    const float sh0 = shift[n], sh1 = shift[n + 1];
    float a0 = 0.f, a1 = 0.f;
    const int rowbase = b * Ll + c * 128;
    for (int r = 0; r < 128; ++r) {
        const int row = rowbase + r;
        const float m = (float)attn[row];
        const f32x2 v = __builtin_nontemporal_load(
            (const f32x2*)(h + (size_t)row * Hn + n));
        a0 += m * fmaxf(v.x * sc0 + sh0, 0.0f);
        a1 += m * fmaxf(v.y * sc1 + sh1, 0.0f);
    }
    f32x2 o; o.x = a0; o.y = a1;
    *(f32x2*)&poolpart[((size_t)c * Bb + b) * Hn + n] = o;
}

// Kernel 3b: fixed-order combine of the 16 chunk partials; divide by count.
__global__ __launch_bounds__(256)
void k_poolsum(const float* __restrict__ poolpart, const float* __restrict__ count,
               float* __restrict__ pooled) {
    const int idx = blockIdx.x * 256 + threadIdx.x;  // 32*512
    const int b = idx >> 9, n = idx & 511;
    float s = 0.f;
    for (int c = 0; c < 16; ++c) s += poolpart[((size_t)c * Bb + b) * Hn + n];
    pooled[idx] = s / fmaxf(count[b], 1.0f);
}

// ---------------------------------------------------------------------------
// Kernel 4: tiny GEMM2 (pool-before-matmul rewrite): out = pooled@W2 + b2
// 32x512x512 = 17 MFLOP -> plain fp32 FMA loop is ample.
// ---------------------------------------------------------------------------
__global__ __launch_bounds__(256)
void k_gemm2(const float* __restrict__ pooled, const float* __restrict__ W2,
             const float* __restrict__ b2, float* __restrict__ out) {
    const int idx = blockIdx.x * 256 + threadIdx.x;  // 16384 outputs
    const int b = idx >> 9, n = idx & 511;
    const float* pb = pooled + (size_t)b * Hn;
    float s = b2[n];
    for (int k = 0; k < Hn; ++k) s = fmaf(pb[k], W2[(size_t)k * Hn + n], s);
    out[idx] = s;
}

// ---------------------------------------------------------------------------
extern "C" void kernel_launch(void* const* d_in, const int* in_sizes, int n_in,
                              void* d_out, int out_size, void* d_ws, size_t ws_size,
                              hipStream_t stream) {
    const float* hidden = (const float*)d_in[0];
    const int*   attn   = (const int*)  d_in[1];
    const float* W1     = (const float*)d_in[2];
    const float* b1     = (const float*)d_in[3];
    const float* gamma  = (const float*)d_in[4];
    const float* beta   = (const float*)d_in[5];
    const float* W2     = (const float*)d_in[6];
    const float* b2     = (const float*)d_in[7];

    float* ws       = (float*)d_ws;
    float* s1part   = ws;                       // 512*512  = 262144
    float* s2part   = ws + 262144;              // 262144
    float* poolpart = ws + 524288;              // 16*32*512 = 262144
    float* pooled   = ws + 786432;              // 16384
    float* scale    = ws + 802816;              // 512
    float* shift    = ws + 803328;              // 512
    float* count    = ws + 803840;              // 32
    float* hbuf     = ws + (1 << 20);           // 65536*512 = 33554432 floats

    k_counts <<<Bb, 256, 0, stream>>>(attn, count);
    k_gemm1  <<<dim3(Mtot / 128, Hn / 128), 256, 0, stream>>>(
        hidden, attn, W1, b1, hbuf, s1part, s2part);
    k_stats  <<<1, 512, 0, stream>>>(s1part, s2part, count, gamma, beta, scale, shift);
    k_pool   <<<dim3(16, Bb), 256, 0, stream>>>(hbuf, attn, scale, shift, poolpart);
    k_poolsum<<<Bb * Hn / 256, 256, 0, stream>>>(poolpart, count, pooled);
    k_gemm2  <<<Bb * Hn / 256, 256, 0, stream>>>(pooled, W2, b2, (float*)d_out);
}
